// ModifiedMLP_6244882448515
// MI455X (gfx1250) — matrix operand order repaired and checked by reading, not compiled, not run on hardware
//
#include <hip/hip_runtime.h>
#include <math.h>

typedef _Float16 v16h __attribute__((ext_vector_type(16)));
typedef _Float16 v8h  __attribute__((ext_vector_type(8)));
typedef float    v8f  __attribute__((ext_vector_type(8)));

#define HIDDEN  50
#define NPAD    64
#define KIN     32     // padded input feature dim (12 real + bias@12 -> 32)
#define KBIAS_IN 12    // K slot carrying the input-layer bias (Xe[12] == 1)
#define KBIAS_Z  50    // K slot carrying the Z-layer bias (staged H[.,50] == 1)
#define ASTRIDE 72     // halves per row of per-wave A staging (144B, 16B aligned)
#define FSTRIDE 68     // floats per row of per-wave final staging (272B, 16B aligned)

// Native v_tanh_f32 (TRANS pipe, co-executes with XDL WMMA) when available;
// otherwise tanh(x) = 1 - 2/(exp(2x)+1) with raw v_exp_f32 / v_rcp_f32.
__device__ __forceinline__ float fast_tanh(float x) {
#if __has_builtin(__builtin_amdgcn_tanhf)
  return __builtin_amdgcn_tanhf(x);
#elif __has_builtin(__builtin_amdgcn_tanh_f32)
  return __builtin_amdgcn_tanh_f32(x);
#else
  float e = __builtin_amdgcn_exp2f(x * 2.8853900817779268f);  // exp(2x)
  float r = __builtin_amdgcn_rcpf(e + 1.0f);
  return fmaf(-2.0f, r, 1.0f);
#endif
}

// A-matrix fragment (16-bit, 16x32 tile): lane L holds row M=L&15.
// lanes 0-15: K = koff+0..15 ; lanes 16-31: K = koff+16..31 (contiguous, matches B).
__device__ __forceinline__ v16h load_fragA(const _Float16* row, int koff, int hi) {
  v8h lo = *(const v8h*)(row + koff + 16 * hi);
  v8h hh = *(const v8h*)(row + koff + 16 * hi + 8);
  return __builtin_shufflevector(lo, hh, 0,1,2,3,4,5,6,7,8,9,10,11,12,13,14,15);
}

// B-matrix fragment (16-bit, 32x16 tile): lane L holds column N=L&15.
// lanes 0-15 hold K = koff+0..15 ; lanes 16-31 hold K = koff+16..31 (contiguous).
__device__ __forceinline__ v16h load_fragB(const _Float16* col, int koff, int hi) {
  v8h lo = *(const v8h*)(col + koff + 16 * hi);
  v8h hh = *(const v8h*)(col + koff + 16 * hi + 8);
  return __builtin_shufflevector(lo, hh, 0,1,2,3,4,5,6,7,8,9,10,11,12,13,14,15);
}

__global__ __launch_bounds__(256) void modified_mlp_wmma(
    const float* __restrict__ X,
    const float* __restrict__ U_W, const float* __restrict__ U_b,
    const float* __restrict__ V_W, const float* __restrict__ V_b,
    const float* __restrict__ H_W, const float* __restrict__ H_b,
    const float* __restrict__ Z_W, const float* __restrict__ Z_b,
    const float* __restrict__ F_W, const float* __restrict__ F_b,
    float* __restrict__ out, int nrows)
{
  // Weights transposed + converted to f16 in LDS: [n (output col)][k] rows.
  // Bias is folded in as an extra K row (multiplied by a constant-1 feature).
  __shared__ __align__(16) _Float16 sWT[3 * NPAD * KIN];   // U,V,H : 64 x 32 each
  __shared__ __align__(16) _Float16 sZT[4 * NPAD * NPAD];  // Z[i]  : 64 x 64
  __shared__ __align__(16) float    sFW[NPAD];
  __shared__ __align__(16) _Float16 sA[8 * 16 * ASTRIDE];  // per-wave A staging (f16)
  __shared__ __align__(16) float    sF[8 * 16 * FSTRIDE];  // per-wave final staging (f32)

  const int tid = threadIdx.x;

  // ---- cooperative weight conversion fp32 -> f16, transposed, bias folded ----
  for (int idx = tid; idx < 3 * NPAD * KIN; idx += 256) {
    int layer = idx / (NPAD * KIN);
    int rem   = idx % (NPAD * KIN);
    int nn = rem / KIN, kk = rem % KIN;
    const float* W = (layer == 0) ? U_W : (layer == 1) ? V_W : H_W;
    const float* B = (layer == 0) ? U_b : (layer == 1) ? V_b : H_b;
    float val = 0.0f;
    if (nn < HIDDEN) {
      if (kk < 12)            val = W[kk * HIDDEN + nn];
      else if (kk == KBIAS_IN) val = B[nn];
    }
    sWT[idx] = (_Float16)val;
  }
  for (int idx = tid; idx < 4 * NPAD * NPAD; idx += 256) {
    int i   = idx >> 12;
    int rem = idx & 4095;
    int nn = rem >> 6, kk = rem & 63;
    float val = 0.0f;
    if (nn < HIDDEN) {
      if (kk < HIDDEN)        val = Z_W[(i * HIDDEN + kk) * HIDDEN + nn];
      else if (kk == KBIAS_Z) val = Z_b[i * HIDDEN + nn];
    }
    sZT[idx] = (_Float16)val;
  }
  for (int idx = tid; idx < NPAD; idx += 256) sFW[idx] = (idx < HIDDEN) ? F_W[idx] : 0.0f;
  __syncthreads();

  const int wave = tid >> 5;
  const int lane = tid & 31;
  const int nl   = lane & 15;     // column (B/C/D) or row (A) index within 16
  const int hi   = lane >> 4;
  const int baseRow = blockIdx.x * 128 + wave * 16;
  _Float16* myA = &sA[wave * 16 * ASTRIDE];
  float*    myF = &sF[wave * 16 * FSTRIDE];

  // ---- Fourier embedding, staged as f16 16x32 A tile (bias-feature at K=12) ----
  {
    int r = baseRow + nl; if (r >= nrows) r = nrows - 1;
    float2 xv = ((const float2*)X)[r];
    float t = xv.x, xs = xv.y;
    _Float16* row = &myA[nl * ASTRIDE];
    if (hi == 0) {
      row[0] = (_Float16)t;
      row[1] = (_Float16)1.0f;
      #pragma unroll
      for (int j = 1; j <= 5; ++j) {
        float a = (float)j * 1.5707963267948966f * xs;   // j * 2*pi/L, L=4
        row[1 + j] = (_Float16)__cosf(a);
        row[6 + j] = (_Float16)__sinf(a);
      }
      row[12] = (_Float16)1.0f;   // constant-1 feature -> bias row of sWT
      #pragma unroll
      for (int k = 13; k < 16; ++k) row[k] = (_Float16)0.0f;
    } else {
      #pragma unroll
      for (int k = 16; k < 32; ++k) row[k] = (_Float16)0.0f;
    }
  }
  __syncthreads();

  v8f u[4], d[4], h[4];   // u = U, d = V - U, h = running hidden state

  // ---- input layers: U, V, H = tanh(Xe @ W) (bias folded into W) ----
  {
    v16h a = load_fragA(&myA[nl * ASTRIDE], 0, hi);
    #pragma unroll
    for (int t4 = 0; t4 < 4; ++t4) {
      int col = 16 * t4 + nl;
      v16h bu = load_fragB(&sWT[0 * NPAD * KIN + col * KIN], 0, hi);
      v16h bv = load_fragB(&sWT[1 * NPAD * KIN + col * KIN], 0, hi);
      v16h bh = load_fragB(&sWT[2 * NPAD * KIN + col * KIN], 0, hi);
      v8f c;

      c = (v8f){};
      c = __builtin_amdgcn_wmma_f32_16x16x32_f16(false, a, false, bu, (short)0, c, false, false);
      #pragma unroll
      for (int e = 0; e < 8; ++e) u[t4][e] = fast_tanh(c[e]);

      c = (v8f){};
      c = __builtin_amdgcn_wmma_f32_16x16x32_f16(false, a, false, bv, (short)0, c, false, false);
      #pragma unroll
      for (int e = 0; e < 8; ++e) d[t4][e] = fast_tanh(c[e]) - u[t4][e];  // V - U

      c = (v8f){};
      c = __builtin_amdgcn_wmma_f32_16x16x32_f16(false, a, false, bh, (short)0, c, false, false);
      #pragma unroll
      for (int e = 0; e < 8; ++e) h[t4][e] = fast_tanh(c[e]);
    }
  }

  // ---- 4 gated Z layers: Z = tanh(H @ Zw) (bias via H col 50 == 1); H = U + Z*(V-U) ----
  #pragma unroll
  for (int i = 0; i < 4; ++i) {
    // stage H as f16 16x64 A tile (C-layout: lane=col nl, rows e+8*hi);
    // padded column KBIAS_Z is forced to 1.0 to activate the bias row of sZT.
    #pragma unroll
    for (int t4 = 0; t4 < 4; ++t4) {
      bool biasCol = (16 * t4 + nl == KBIAS_Z);
      #pragma unroll
      for (int e = 0; e < 8; ++e) {
        _Float16 hv = biasCol ? (_Float16)1.0f : (_Float16)h[t4][e];
        myA[(e + 8 * hi) * ASTRIDE + 16 * t4 + nl] = hv;
      }
    }
    __syncthreads();

    v16h a0 = load_fragA(&myA[nl * ASTRIDE], 0, hi);
    v16h a1 = load_fragA(&myA[nl * ASTRIDE], 32, hi);
    #pragma unroll
    for (int t4 = 0; t4 < 4; ++t4) {
      int col = 16 * t4 + nl;
      const _Float16* zcol = &sZT[i * NPAD * NPAD + col * NPAD];
      v16h b0 = load_fragB(zcol, 0, hi);
      v16h b1 = load_fragB(zcol, 32, hi);
      v8f c = (v8f){};
      c = __builtin_amdgcn_wmma_f32_16x16x32_f16(false, a0, false, b0, (short)0, c, false, false);
      c = __builtin_amdgcn_wmma_f32_16x16x32_f16(false, a1, false, b1, (short)0, c, false, false);
      #pragma unroll
      for (int e = 0; e < 8; ++e) {
        float z = fast_tanh(c[e]);
        h[t4][e] = fmaf(z, d[t4][e], u[t4][e]);   // U + Z*(V-U)
      }
    }
    __syncthreads();
  }

  // ---- final: out = H @ F_W + F_b (dot per row via fp32 LDS transpose) ----
  #pragma unroll
  for (int t4 = 0; t4 < 4; ++t4) {
    #pragma unroll
    for (int e = 0; e < 8; ++e)
      myF[(e + 8 * hi) * FSTRIDE + 16 * t4 + nl] = h[t4][e];
  }
  __syncthreads();

  if (hi == 0) {
    int r = baseRow + nl;
    if (r < nrows) {
      const float* rowp = &myF[nl * FSTRIDE];
      float acc = F_b[0];
      #pragma unroll
      for (int k = 0; k < HIDDEN; ++k) acc = fmaf(rowp[k], sFW[k], acc);
      out[r] = acc;
    }
  }
}

extern "C" void kernel_launch(void* const* d_in, const int* in_sizes, int n_in,
                              void* d_out, int out_size, void* d_ws, size_t ws_size,
                              hipStream_t stream) {
  (void)d_ws; (void)ws_size; (void)n_in; (void)out_size;
  const float* X   = (const float*)d_in[0];
  const float* U_W = (const float*)d_in[1];
  const float* U_b = (const float*)d_in[2];
  const float* V_W = (const float*)d_in[3];
  const float* V_b = (const float*)d_in[4];
  const float* H_W = (const float*)d_in[5];
  const float* H_b = (const float*)d_in[6];
  const float* Z_W = (const float*)d_in[7];
  const float* Z_b = (const float*)d_in[8];
  const float* F_W = (const float*)d_in[9];
  const float* F_b = (const float*)d_in[10];
  float* out = (float*)d_out;

  int nrows = in_sizes[0] / 2;           // X is (N, 2)
  int grid  = (nrows + 127) / 128;       // 128 rows per block (8 waves x 16 rows)
  modified_mlp_wmma<<<grid, 256, 0, stream>>>(
      X, U_W, U_b, V_W, V_b, H_W, H_b, Z_W, Z_b, F_W, F_b, out, nrows);
}
